// AttentionHeader_87952340288224
// MI455X (gfx1250) — matrix operand order repaired and checked by reading, not compiled, not run on hardware
//
#include <hip/hip_runtime.h>
#include <hip/hip_bf16.h>

// ---------------------------------------------------------------------------
// GAT attention head, MI455X (gfx1250) fused implementation.
// logits[i,j] = f1[i] + f2[j]  (rank-structured: no QK^T GEMM needed)
// exp factorization: p_ij = (z>0) ? A_hi[i]*E_hi[j] : A_lo[i]*E_lo[j]
//   -> all transcendentals hoisted out of the O(N^2) loop.
// Heavy op = coefs @ seq_fts : v_wmma_f32_16x16x32_bf16, fp32 accumulate.
// Softmax denominators computed as P * ones via a 5th WMMA accumulator.
// eh/el tables staged into LDS with the Tensor Data Mover when available.
// ---------------------------------------------------------------------------

typedef __attribute__((ext_vector_type(16))) __bf16 v16bf;
typedef __attribute__((ext_vector_type(8)))  float  v8f;
typedef unsigned int u32x4 __attribute__((ext_vector_type(4)));
typedef int          i32x8 __attribute__((ext_vector_type(8)));
typedef int          i32x4 __attribute__((ext_vector_type(4)));

union V16U { v16bf v; unsigned int u[8]; };

#define GN 8192   // nodes
#define GF 256    // input features
#define GD 64     // hidden dim

__device__ __forceinline__ unsigned short f2bf(float x) {
    unsigned int uu = __float_as_uint(x);
    unsigned int r  = uu + 0x7FFFu + ((uu >> 16) & 1u);   // round-to-nearest-even
    return (unsigned short)(r >> 16);
}

// ---------------------------------------------------------------------------
// TDM helper: copy a contiguous run of `bytes/4` dwords from global to LDS.
// D# per CDNA5 ISA ch.8: group0 = {count/flags, lds_addr, global_addr, type=2},
// group1 = {data_size=4B, tensor_dim0=tile_dim0=n, tensor_dim1=tile_dim1=1,
//           tensor_dim0_stride=n}, groups 2/3 zero (tensor <= 2D).
// ---------------------------------------------------------------------------
#if __has_builtin(__builtin_amdgcn_tensor_load_to_lds)
#define HAVE_TDM 1
__device__ __forceinline__ void tdm_load_row(void* lds_dst, const void* global_src,
                                             unsigned n_dwords) {
    unsigned long long ga = (unsigned long long)(uintptr_t)global_src;
    unsigned lds_off = (unsigned)(uintptr_t)lds_dst;        // LDS byte offset: low 32 bits of the pointer
    u32x4 g0;
    g0[0] = 1u;                                             // count=1, user descriptor
    g0[1] = lds_off;                                        // lds_addr (bytes)
    g0[2] = (unsigned)(ga & 0xffffffffu);                   // global_addr[31:0]
    g0[3] = (unsigned)((ga >> 32) & 0x1ffffffu) | (2u << 30); // addr[56:32] | type=2
    i32x8 g1;
    g1[0] = 2 << 16;                                        // data_size = 4 bytes
    g1[1] = (int)((n_dwords & 0xffffu) << 16);              // tensor_dim0[15:0]
    g1[2] = (int)((n_dwords >> 16) & 0xffffu) | (1 << 16);  // tensor_dim0[31:16], tensor_dim1=1
    g1[3] = (int)((n_dwords & 0xffffu) << 16);              // tile_dim0
    g1[4] = 1;                                              // tile_dim1 = 1
    g1[5] = (int)n_dwords;                                  // tensor_dim0_stride[31:0]
    g1[6] = 0;
    g1[7] = 0;
    i32x4 gz = {0, 0, 0, 0};
#if __clang_major__ >= 23
    i32x8 gz8 = {0, 0, 0, 0, 0, 0, 0, 0};
    __builtin_amdgcn_tensor_load_to_lds(g0, g1, gz, gz, gz8, 0);
#else
    __builtin_amdgcn_tensor_load_to_lds(g0, g1, gz, gz, 0);
#endif
}
#else
#define HAVE_TDM 0
#endif

// ---------------------------------------------------------------------------
// Kernel 0: W0 [256,64] f32  ->  W0T [64][256] bf16  (B-operand friendly layout)
// ---------------------------------------------------------------------------
__global__ void w0t_kernel(const float* __restrict__ W0, unsigned short* __restrict__ w0t) {
    int idx = blockIdx.x * blockDim.x + threadIdx.x;      // 64*256 threads
    int d = idx >> 8;          // 0..63
    int k = idx & 255;         // 0..255
    w0t[d * GF + k] = f2bf(W0[k * GD + d]);
}

// ---------------------------------------------------------------------------
// Kernel A: projection  seq_fts = seq @ W0   (+ f1, f2 side outputs)
// One wave per 16-row tile, full D=64 (4 C-tiles), K=256 in 8 steps of 32.
// ---------------------------------------------------------------------------
#define WAVES_A 2
__global__ void proj_kernel(const float* __restrict__ seq,
                            const unsigned short* __restrict__ w0t,
                            const float* __restrict__ w1, const float* __restrict__ b1,
                            const float* __restrict__ w2, const float* __restrict__ b2,
                            unsigned short* __restrict__ sfT,
                            float* __restrict__ f1, float* __restrict__ f2) {
    __shared__ __align__(32) unsigned short s_t[WAVES_A][GD * 16];

    const int w    = threadIdx.x >> 5;
    const int lane = threadIdx.x & 31;
    const int ml   = lane & 15;
    const int hi   = lane >> 4;
    const int i0   = (blockIdx.x * WAVES_A + w) * 16;

    v8f acc[4] = {};

    const float* sp = seq + (size_t)(i0 + ml) * GF;

    for (int k0 = 0; k0 < GF; k0 += 32) {
        // A operand: 16x32 bf16 tile of seq. Lane layout (matches B):
        // lanes 0-15 hold K {0..15}, lanes 16-31 hold K {16..31}.
        float4 a0 = *(const float4*)(sp + k0 + hi * 16);
        float4 a1 = *(const float4*)(sp + k0 + hi * 16 + 4);
        float4 a2 = *(const float4*)(sp + k0 + hi * 16 + 8);
        float4 a3 = *(const float4*)(sp + k0 + hi * 16 + 12);
        float as[16] = {a0.x,a0.y,a0.z,a0.w, a1.x,a1.y,a1.z,a1.w,
                        a2.x,a2.y,a2.z,a2.w, a3.x,a3.y,a3.z,a3.w};
        V16U av;
        #pragma unroll
        for (int i = 0; i < 16; i += 2)
            av.u[i >> 1] = __builtin_amdgcn_perm(__float_as_uint(as[i + 1]),
                                                 __float_as_uint(as[i]), 0x07060302u);

        #pragma unroll
        for (int t = 0; t < 4; ++t) {
            // B operand: lane = D-column, 16 contiguous K values (32B aligned).
            const v16bf b = *(const v16bf*)(w0t + (size_t)(t * 16 + ml) * GF + k0 + hi * 16);
            acc[t] = __builtin_amdgcn_wmma_f32_16x16x32_bf16(
                false, av.v, false, b, (short)0, acc[t], false, false);
        }
    }

    // f1/f2 from C-layout accumulators: lane holds rows r+8*hi, col t*16+ml.
    float p1[8] = {}, p2[8] = {};
    #pragma unroll
    for (int t = 0; t < 4; ++t) {
        float w1v = w1[t * 16 + ml];
        float w2v = w2[t * 16 + ml];
        #pragma unroll
        for (int r = 0; r < 8; ++r) {
            p1[r] += acc[t][r] * w1v;
            p2[r] += acc[t][r] * w2v;
        }
    }
    #pragma unroll
    for (int off = 1; off < 16; off <<= 1) {
        #pragma unroll
        for (int r = 0; r < 8; ++r) {
            p1[r] += __shfl_xor(p1[r], off, 32);
            p2[r] += __shfl_xor(p2[r], off, 32);
        }
    }
    if (ml == 0) {
        float b1v = b1[0], b2v = b2[0];
        #pragma unroll
        for (int r = 0; r < 8; ++r) {
            f1[i0 + r + hi * 8] = p1[r] + b1v;
            f2[i0 + r + hi * 8] = p2[r] + b2v;
        }
    }

    // Stage tile through LDS as [d][i_local] bf16, then coalesced store to sfT.
    #pragma unroll
    for (int t = 0; t < 4; ++t)
        #pragma unroll
        for (int r = 0; r < 8; ++r)
            s_t[w][(t * 16 + ml) * 16 + r + hi * 8] = f2bf(acc[t][r]);
    __syncthreads();

    for (int d = lane; d < GD; d += 32) {
        const uint4* lp = (const uint4*)&s_t[w][d * 16];
        uint4 x0 = lp[0], x1 = lp[1];
        uint4* gp = (uint4*)(sfT + (size_t)d * GN + i0);
        gp[0] = x0; gp[1] = x1;
    }
}

// ---------------------------------------------------------------------------
// Kernel B: f2max = max_j f2[j], then E_hi[j] = exp(f2j - f2max),
//           E_lo[j] = exp(0.2*(f2j - f2max)).  All <= 1, no overflow.
// ---------------------------------------------------------------------------
__global__ void prep_kernel(const float* __restrict__ f2, float* __restrict__ f2max,
                            float* __restrict__ eh, float* __restrict__ el) {
    __shared__ float s[256];
    float m = -3.402823466e38f;
    for (int i = threadIdx.x; i < GN; i += 256) m = fmaxf(m, f2[i]);
    s[threadIdx.x] = m;
    __syncthreads();
    for (int off = 128; off > 0; off >>= 1) {
        if ((int)threadIdx.x < off) s[threadIdx.x] = fmaxf(s[threadIdx.x], s[threadIdx.x + off]);
        __syncthreads();
    }
    m = s[0];
    if (threadIdx.x == 0) *f2max = m;
    for (int i = threadIdx.x; i < GN; i += 256) {
        float d = f2[i] - m;
        eh[i] = __expf(d);
        el[i] = __expf(0.2f * d);
    }
}

// ---------------------------------------------------------------------------
// Kernel C: fused select -> scale -> (coefs @ seq_fts) -> /rowsum + bias
// One wave per 16-row tile; j swept in steps of 32 (WMMA K).
// Hot loop: cmp/cndmask/mul + v_perm pack + 5 WMMA (4 value tiles + row sums).
// ---------------------------------------------------------------------------
#define WAVES_C 2
__global__ void attn_kernel(const unsigned short* __restrict__ sfT,
                            const float* __restrict__ f1,
                            const float* __restrict__ eh_g,
                            const float* __restrict__ el_g,
                            const float* __restrict__ f2maxp,
                            const float* __restrict__ bias,
                            float* __restrict__ out) {
    __shared__ __align__(32) float s_eh[GN];        // 32 KB
    __shared__ __align__(32) float s_el[GN];        // 32 KB

    const int tid = threadIdx.x;
#if HAVE_TDM
    if (tid < 32) {                                 // wave 0 drives the TDM
        tdm_load_row(s_eh, eh_g, GN);
        tdm_load_row(s_el, el_g, GN);
        __builtin_amdgcn_s_wait_tensorcnt(0);
    }
#else
    for (int i = tid; i < GN / 4; i += WAVES_C * 32) {
        ((float4*)s_eh)[i] = ((const float4*)eh_g)[i];
        ((float4*)s_el)[i] = ((const float4*)el_g)[i];
    }
#endif
    __syncthreads();

    const int w    = tid >> 5;
    const int lane = tid & 31;
    const int ml   = lane & 15;
    const int hi   = lane >> 4;
    const int i0   = (blockIdx.x * WAVES_C + w) * 16;

    const float f1v = f1[i0 + ml];                   // A-matrix row for this lane
    const float fm  = *f2maxp;
    const float mz  = f1v + fm;
    const float mrow = mz > 0.f ? mz : 0.2f * mz;    // lrelu(f1+f2max): exact row max
    const float A_hi = __expf(mz - mrow);            // <= 1
    const float A_lo = __expf(0.2f * mz - mrow);     // <= 1
    const float th   = __expf(-mz);                  // E_hi[j] > th  <=>  f1+f2j > 0

    V16U ones;                                       // all-ones bf16 B matrix
    #pragma unroll
    for (int i = 0; i < 8; ++i) ones.u[i] = 0x3F803F80u;

    v8f acc[4] = {};
    v8f acc5 = {};                                   // P * ones = row sums
    const int koff0 = hi * 16;
    const int koff1 = hi * 16 + 8;

    for (int j0 = 0; j0 < GN; j0 += 32) {
        const float4* qh0 = (const float4*)&s_eh[j0 + koff0];
        const float4* qh1 = (const float4*)&s_eh[j0 + koff1];
        const float4* ql0 = (const float4*)&s_el[j0 + koff0];
        const float4* ql1 = (const float4*)&s_el[j0 + koff1];
        float4 h0 = qh0[0], h1 = qh0[1], h2 = qh1[0], h3 = qh1[1];
        float4 l0 = ql0[0], l1 = ql0[1], l2 = ql1[0], l3 = ql1[1];
        float eh[16] = {h0.x,h0.y,h0.z,h0.w, h1.x,h1.y,h1.z,h1.w,
                        h2.x,h2.y,h2.z,h2.w, h3.x,h3.y,h3.z,h3.w};
        float el[16] = {l0.x,l0.y,l0.z,l0.w, l1.x,l1.y,l1.z,l1.w,
                        l2.x,l2.y,l2.z,l2.w, l3.x,l3.y,l3.z,l3.w};

        V16U av;
        #pragma unroll
        for (int i = 0; i < 16; i += 2) {
            bool  c0 = eh[i]     > th;
            bool  c1 = eh[i + 1] > th;
            float p0 = (c0 ? A_hi : A_lo) * (c0 ? eh[i]     : el[i]);
            float p1 = (c1 ? A_hi : A_lo) * (c1 ? eh[i + 1] : el[i + 1]);
            // pack two bf16 (truncated f32 high halves) with one v_perm_b32
            av.u[i >> 1] = __builtin_amdgcn_perm(__float_as_uint(p1),
                                                 __float_as_uint(p0), 0x07060302u);
        }

        #pragma unroll
        for (int t = 0; t < 4; ++t) {
            const unsigned short* bp = sfT + (size_t)(t * 16 + ml) * GN + j0 + hi * 16;
            v16bf b = *(const v16bf*)bp;
            __builtin_prefetch(bp + 32, 0, 1);       // next j-step (global_prefetch_b8)
            acc[t] = __builtin_amdgcn_wmma_f32_16x16x32_bf16(
                false, av.v, false, b, (short)0, acc[t], false, false);
        }
        acc5 = __builtin_amdgcn_wmma_f32_16x16x32_bf16(
            false, av.v, false, ones.v, (short)0, acc5, false, false);
    }

    // acc5[r] (any column) = softmax denominator for row i0 + r + 8*hi.
    #pragma unroll
    for (int r = 0; r < 8; ++r) {
        float inv = 1.0f / acc5[r];
        #pragma unroll
        for (int t = 0; t < 4; ++t) {
            float bv = bias[t * 16 + ml];
            out[(size_t)(i0 + r + hi * 8) * GD + t * 16 + ml] = acc[t][r] * inv + bv;
        }
    }
}

// ---------------------------------------------------------------------------
extern "C" void kernel_launch(void* const* d_in, const int* in_sizes, int n_in,
                              void* d_out, int out_size, void* d_ws, size_t ws_size,
                              hipStream_t stream) {
    const float* seq  = (const float*)d_in[0];
    const float* W0   = (const float*)d_in[1];
    const float* w1   = (const float*)d_in[2];
    const float* b1   = (const float*)d_in[3];
    const float* w2   = (const float*)d_in[4];
    const float* b2   = (const float*)d_in[5];
    const float* bias = (const float*)d_in[6];
    float* out = (float*)d_out;

    char* ws = (char*)d_ws;
    unsigned short* w0t = (unsigned short*)ws;                       //   32 KB
    unsigned short* sfT = (unsigned short*)(ws + 32768);             // 1024 KB, bf16 [64][8192]
    float* f1  = (float*)(ws + 32768 + 1048576);                     //   32 KB
    float* f2  = f1 + GN;                                            //   32 KB
    float* eh  = f2 + GN;                                            //   32 KB
    float* el  = eh + GN;                                            //   32 KB
    float* f2m = el + GN;                                            //    4 B

    w0t_kernel <<<64, 256, 0, stream>>>(W0, w0t);
    proj_kernel<<<GN / 16 / WAVES_A, WAVES_A * 32, 0, stream>>>(seq, w0t, w1, b1, w2, b2,
                                                                sfT, f1, f2);
    prep_kernel<<<1, 256, 0, stream>>>(f2, f2m, eh, el);
    attn_kernel<<<GN / 16 / WAVES_C, WAVES_C * 32, 0, stream>>>(sfT, f1, eh, el, f2m, bias, out);
}
